// DualReconstruction_80290118632221
// MI455X (gfx1250) — matrix operand order repaired and checked by reading, not compiled, not run on hardware
//
#include <hip/hip_runtime.h>
#include <math.h>

// ---------------------------------------------------------------------------
// DualReconstruction on gfx1250 (MI455X), wave32 + WMMA bf16 + TDM.
//
// Shapes: query (32,196,384) f32, support (25,196,384) f32.
// Outputs (flat, in order): recon_q (32,196,384), recon_s (25,196,384),
//   q_loss, s_loss, total, attn_q (32,25,196,196), attn_s (25,32,196,196).
// ---------------------------------------------------------------------------

typedef __attribute__((ext_vector_type(16))) __bf16 v16bf;
typedef __attribute__((ext_vector_type(8)))  __bf16 v8bf;
typedef __attribute__((ext_vector_type(8)))  float  v8f;
typedef __attribute__((ext_vector_type(4)))  unsigned int v4u;
typedef __attribute__((ext_vector_type(8)))  int    v8i;
typedef __attribute__((ext_vector_type(4)))  int    v4i;

union AFrag { v16bf v; v8bf h[2]; };

__device__ __forceinline__ v8f wmma_bf16(v16bf a, v16bf b, v8f c) {
    // D = A(16x32 bf16) x B(32x16 bf16) + C(16x16 f32)
    return __builtin_amdgcn_wmma_f32_16x16x32_bf16(
        /*neg_a=*/false, a, /*neg_b=*/false, b,
        /*c_mod=*/(short)0, c, /*reuse_a=*/false, /*reuse_b=*/false);
}

static constexpr int LEN = 196;   // sequence length
static constexpr int CH  = 384;   // channels
static constexpr int LP  = 224;   // K-padded length for recon GEMM (7 x 32)
static constexpr int NQ  = 32;    // #query items
static constexpr int NS  = 25;    // #support items
static constexpr int ROWS_Q = NQ * LEN;   // 6272
static constexpr int ROWS_S = NS * LEN;   // 4900
static constexpr int JT  = 13;    // ceil(196/16) column tiles
static constexpr int KC_SIM = CH / 32;    // 12 K-chunks for sim GEMM
static constexpr int KC_REC = LP / 32;    // 7 K-chunks for recon GEMM

// ---------------------------------------------------------------------------
// TDM: load a 2D bf16 tile [16 x 384] (row-major, stride 384) into LDS.
// Rows past `rem_rows` are hardware zero-filled (OOB reads return zero).
// D# bit packing per CDNA5 ISA ch.8 (groups 0/1; 2D -> groups 2/3 zero).
// ---------------------------------------------------------------------------
__device__ __forceinline__ void tdm_load_A(const __bf16* gsrc, __bf16* lds_dst,
                                           int rem_rows)
{
    unsigned long long ga = (unsigned long long)(uintptr_t)gsrc;   // byte address
    v4u g0;
    g0[0] = 1u;                                    // count=1 (valid), no gather
    g0[1] = (unsigned)(uintptr_t)lds_dst;          // lds_addr (byte offset)
    g0[2] = (unsigned)ga;                          // global_addr[31:0]
    g0[3] = ((unsigned)(ga >> 32) & 0x01FFFFFFu)   // global_addr[56:32]
            | (2u << 30);                          // type = 2 ("image")
    v8i g1;
    g1[0] = 1 << 16;          // workgroup_mask=0 | data_size=1 (2 bytes)
    g1[1] = CH << 16;         // tensor_dim0[15:0] = 384
    g1[2] = rem_rows << 16;   // tensor_dim1[15:0] = remaining rows (zero-fill past)
    g1[3] = CH << 16;         // tile_dim0 = 384
    g1[4] = 16;               // tile_dim1 = 16, tile_dim2 = 0 (2D)
    g1[5] = CH;               // tensor_dim0_stride = 384
    g1[6] = 0;
    g1[7] = 0;
    v4i g2 = {0, 0, 0, 0};
    v4i g3 = {0, 0, 0, 0};
#if defined(__clang_major__) && (__clang_major__ >= 23)
    v8i gz = {0, 0, 0, 0, 0, 0, 0, 0};
    __builtin_amdgcn_tensor_load_to_lds(g0, g1, g2, g3, gz, 0);
#else
    __builtin_amdgcn_tensor_load_to_lds(g0, g1, g2, g3, 0);
#endif
}

// ---------------------------------------------------------------------------
// L2-normalize rows of both tensors, cast to bf16 (wave per row).
// ---------------------------------------------------------------------------
__global__ __launch_bounds__(256)
void norm_cast_kernel(const float* __restrict__ q, const float* __restrict__ s,
                      __bf16* __restrict__ qn, __bf16* __restrict__ sn)
{
    int row  = blockIdx.x * 8 + (threadIdx.x >> 5);
    int lane = threadIdx.x & 31;
    if (row >= ROWS_Q + ROWS_S) return;
    const float* src; __bf16* dst;
    if (row < ROWS_Q) { src = q + (size_t)row * CH;            dst = qn + (size_t)row * CH; }
    else { int r = row - ROWS_Q; src = s + (size_t)r * CH;     dst = sn + (size_t)r * CH; }

    float vals[12];
    float sq = 0.f;
#pragma unroll
    for (int i = 0; i < 12; ++i) { vals[i] = src[lane + 32 * i]; sq += vals[i] * vals[i]; }
#pragma unroll
    for (int off = 16; off; off >>= 1) sq += __shfl_xor(sq, off);
    float inv = 1.f / fmaxf(sqrtf(sq), 1e-8f);
#pragma unroll
    for (int i = 0; i < 12; ++i) dst[lane + 32 * i] = (__bf16)(vals[i] * inv);
}

// ---------------------------------------------------------------------------
// Build pre-transposed, K-padded bf16 value copies:
//   xT[item][c][j] = (j < 196) ? x[item][j][c] : 0   with j-extent padded to 224.
// This makes the recon GEMM's WMMA B-fragments single aligned 32B loads.
// ---------------------------------------------------------------------------
__global__ __launch_bounds__(256)
void transpose_cast_kernel(const float* __restrict__ q, const float* __restrict__ s,
                           __bf16* __restrict__ qT, __bf16* __restrict__ sT)
{
    constexpr int TOT = (NQ + NS) * CH * LP;
    int idx = blockIdx.x * 256 + threadIdx.x;
    if (idx >= TOT) return;
    int j    = idx % LP;
    int rest = idx / LP;
    int c    = rest % CH;
    int item = rest / CH;
    float v = 0.f;
    if (j < LEN) {
        v = (item < NQ) ? q[((size_t)item * LEN + j) * CH + c]
                        : s[((size_t)(item - NQ) * LEN + j) * CH + c];
    }
    __bf16 bv = (__bf16)v;
    if (item < NQ) qT[idx] = bv;
    else           sT[idx - NQ * CH * LP] = bv;
}

// ---------------------------------------------------------------------------
// Fused cross-attention: per block = one (a,b) pair x one 16-row slab.
//   Phase 0: TDM loads A slab (16x384 bf16) into LDS, zero-filled past row 195.
//   Phase 1: sim(16x196) = A . Bn[b]^T via WMMA bf16 (K=384), B software-pipelined.
//   Phase 2: row softmax -> attn_out (nontemporal stream), P -> bf16 LDS.
//   Phase 3: recon partial (16x384) = P(16x224) . V(224x384) via WMMA bf16,
//            B-fragments read directly from pre-transposed VT in global.
//   Phase 4: f32 atomics accumulate into recon_out[a].
// ---------------------------------------------------------------------------
__global__ __launch_bounds__(128)
void cross_attn_kernel(const __bf16* __restrict__ An,
                       const __bf16* __restrict__ Bn,
                       const __bf16* __restrict__ VT,
                       float* __restrict__ attn_out,
                       float* __restrict__ recon_out,
                       int NB)
{
    __shared__ __align__(32) __bf16 lds_A[16 * CH];      // 12 KB  A slab (TDM target)
    __shared__ __align__(32) __bf16 lds_P[16 * LP];      //  7 KB  probs, K-padded
    __shared__ float lds_S[16 * 208];                    // 13 KB  sim row-slab

    const int tid   = threadIdx.x;
    const int wave  = tid >> 5;
    const int lane  = tid & 31;
    const int half  = lane >> 4;   // K-half (A/B frag) / M-half (C frag)
    const int nlane = lane & 15;   // row (A) / column (B,C) within tile

    const int pair = blockIdx.x;
    const int a    = pair / NB;
    const int bidx = pair - a * NB;
    const int i0   = blockIdx.y * 16;

    const __bf16* Arow0 = An + (size_t)a * LEN * CH;
    const __bf16* Brow0 = Bn + (size_t)bidx * LEN * CH;
    const __bf16* VTb   = VT + (size_t)bidx * CH * LP;

    // warm L2 for this wave's first B tile (global_prefetch_b8)
    __builtin_prefetch(Brow0 + (size_t)(wave * 16) * CH, 0, 1);

    // ---- Phase 0: async tensor DMA of the A slab into LDS ----
    if (wave == 0) {
        tdm_load_A(Arow0 + (size_t)i0 * CH, lds_A, LEN - i0);
        __builtin_amdgcn_s_wait_tensorcnt(0);
    }
    __syncthreads();

    // ---- Phase 1: sim tiles, one 16x16 tile per wave-iteration ----
    for (int jt = wave; jt < JT; jt += 4) {
        int j0 = jt * 16;
        int jr = j0 + nlane; if (jr > LEN - 1) jr = LEN - 1;
        // B operand: lane n holds column n == row (j0+n) of Bn; K-half by lane group.
        const __bf16* Bp = Brow0 + (size_t)jr * CH + half * 16;
        v8f acc = {};
        v16bf bcur = *(const v16bf*)(Bp);            // software-pipelined B fragment
#pragma unroll
        for (int kc = 0; kc < KC_SIM; ++kc) {
            v16bf bnext;
            if (kc + 1 < KC_SIM) bnext = *(const v16bf*)(Bp + (kc + 1) * 32);
            AFrag af;   // lane m: row m, K = kc*32 + half*16 + {0..15}
            const __bf16* ap = &lds_A[nlane * CH + kc * 32 + half * 16];
            af.h[0] = *(const v8bf*)ap;
            af.h[1] = *(const v8bf*)(ap + 8);
            acc = wmma_bf16(af.v, bcur, acc);
            bcur = bnext;
        }
#pragma unroll
        for (int v = 0; v < 8; ++v)          // C frag: VGPR v <-> M = v + 8*half, N = nlane
            lds_S[(v + 8 * half) * 208 + j0 + nlane] = acc[v];
    }
    __syncthreads();

    // ---- Phase 2: row softmax (8 lanes per row), stream attn, build P bf16 ----
    {
        int row = tid >> 3;
        int t   = tid & 7;
        const float* srow = &lds_S[row * 208];
        float m = -1e30f;
        for (int j = t; j < LEN; j += 8) m = fmaxf(m, srow[j]);
        m = fmaxf(m, __shfl_xor(m, 1));
        m = fmaxf(m, __shfl_xor(m, 2));
        m = fmaxf(m, __shfl_xor(m, 4));
        float ssum = 0.f;
        for (int j = t; j < LEN; j += 8) ssum += __expf(srow[j] - m);
        ssum += __shfl_xor(ssum, 1);
        ssum += __shfl_xor(ssum, 2);
        ssum += __shfl_xor(ssum, 4);
        float inv = 1.f / ssum;
        int i_row = i0 + row;
        float* arow = attn_out + ((size_t)pair * LEN + i_row) * LEN;
        for (int j = t; j < LP; j += 8) {
            float p = 0.f;
            if (j < LEN) {
                p = __expf(srow[j] - m) * inv;
                if (i_row < LEN) __builtin_nontemporal_store(p, arow + j);
            }
            lds_P[row * LP + j] = (__bf16)p;   // zero pad for K=196..223
        }
    }
    __syncthreads();

    // ---- Phase 3: recon partial = P(16x224) x V(224x384), B direct from VT ----
    v8f racc[6] = {};
#pragma unroll
    for (int kc = 0; kc < KC_REC; ++kc) {
        AFrag pf;                              // A operand from P (LDS)
        const __bf16* pp = &lds_P[nlane * LP + kc * 32 + half * 16];
        pf.h[0] = *(const v8bf*)pp;
        pf.h[1] = *(const v8bf*)(pp + 8);
#pragma unroll
        for (int t = 0; t < 6; ++t) {          // 4 waves x 6 tiles = 384 output cols
            int col = (wave * 6 + t) * 16 + nlane;
            // B operand: lane n holds column col of V == VT row col; K-half by lane group.
            v16bf vf = *(const v16bf*)(VTb + (size_t)col * LP + kc * 32 + half * 16);
            racc[t] = wmma_bf16(pf.v, vf, racc[t]);
        }
    }

    // ---- Phase 4: accumulate recon over pairs via f32 atomics ----
#pragma unroll
    for (int t = 0; t < 6; ++t) {
        int col = (wave * 6 + t) * 16 + nlane;
#pragma unroll
        for (int v = 0; v < 8; ++v) {
            int i_row = i0 + v + 8 * half;
            if (i_row < LEN) {
                __hip_atomic_fetch_add(&recon_out[((size_t)a * LEN + i_row) * CH + col],
                                       racc[t][v], __ATOMIC_RELAXED, __HIP_MEMORY_SCOPE_AGENT);
            }
        }
    }
}

// ---------------------------------------------------------------------------
// Cosine-similarity reduction (wave per row) into 2 accumulators.
// ---------------------------------------------------------------------------
__global__ __launch_bounds__(256)
void cos_loss_kernel(const float* __restrict__ q, const float* __restrict__ s,
                     const float* __restrict__ rq, const float* __restrict__ rs,
                     float* __restrict__ acc)
{
    int row  = blockIdx.x * 8 + (threadIdx.x >> 5);
    int lane = threadIdx.x & 31;
    if (row >= ROWS_Q + ROWS_S) return;
    const float *o, *r; float* ap;
    if (row < ROWS_Q) { o = q + (size_t)row * CH; r = rq + (size_t)row * CH; ap = acc; }
    else { int r2 = row - ROWS_Q; o = s + (size_t)r2 * CH; r = rs + (size_t)r2 * CH; ap = acc + 1; }

    float oo = 0.f, rr = 0.f, dot = 0.f;
#pragma unroll
    for (int i = 0; i < 12; ++i) {
        float x = o[lane + 32 * i], y = r[lane + 32 * i];
        oo += x * x; rr += y * y; dot += x * y;
    }
#pragma unroll
    for (int off = 16; off; off >>= 1) {
        oo += __shfl_xor(oo, off); rr += __shfl_xor(rr, off); dot += __shfl_xor(dot, off);
    }
    if (lane == 0) {
        float v = dot / (fmaxf(sqrtf(oo), 1e-8f) * fmaxf(sqrtf(rr), 1e-8f));
        __hip_atomic_fetch_add(ap, v, __ATOMIC_RELAXED, __HIP_MEMORY_SCOPE_AGENT);
    }
}

__global__ void finalize_kernel(const float* __restrict__ acc, float* __restrict__ losses)
{
    if (threadIdx.x == 0) {
        float ql = 1.f - acc[0] / (float)ROWS_Q;
        float sl = 1.f - acc[1] / (float)ROWS_S;
        losses[0] = ql; losses[1] = sl; losses[2] = ql + sl;
    }
}

__global__ void zero_kernel(float* __restrict__ p, size_t n)
{
    size_t i = (size_t)blockIdx.x * blockDim.x + threadIdx.x;
    if (i < n) p[i] = 0.f;
}

// ---------------------------------------------------------------------------
extern "C" void kernel_launch(void* const* d_in, const int* in_sizes, int n_in,
                              void* d_out, int out_size, void* d_ws, size_t ws_size,
                              hipStream_t stream)
{
    (void)in_sizes; (void)n_in; (void)out_size; (void)ws_size;
    const float* q = (const float*)d_in[0];   // (32,196,384)
    const float* s = (const float*)d_in[1];   // (25,196,384)

    float* out     = (float*)d_out;
    float* recon_q = out;                               // 2,408,448
    float* recon_s = out + 2408448;                     // 1,881,600
    float* losses  = out + 2408448 + 1881600;           // 3
    float* attn_q  = losses + 3;                        // 30,732,800
    float* attn_s  = attn_q + (size_t)30732800;         // 30,732,800

    // workspace layout (all segments 256B-aligned):
    //   qn (norm bf16)  4,816,896 B
    //   sn (norm bf16)  3,763,200 B
    //   qT (val bf16, transposed, K-padded to 224)  5,505,024 B
    //   sT (val bf16, transposed, K-padded to 224)  4,300,800 B
    //   accs (2 x f32)
    char* ws = (char*)d_ws;
    __bf16* qn = (__bf16*)ws;
    __bf16* sn = (__bf16*)(ws + 4816896);
    __bf16* qT = (__bf16*)(ws + 4816896 + 3763200);
    __bf16* sT = (__bf16*)(ws + 4816896 + 3763200 + 5505024);
    float* accs = (float*)(ws + 4816896 + 3763200 + 5505024 + 4300800);

    // recon buffers are atomic accumulation targets -> zero every call
    size_t nz = (size_t)(2408448 + 1881600);
    zero_kernel<<<(unsigned)((nz + 255) / 256), 256, 0, stream>>>(out, nz);
    zero_kernel<<<1, 32, 0, stream>>>(accs, 2);

    norm_cast_kernel<<<(ROWS_Q + ROWS_S + 7) / 8, 256, 0, stream>>>(q, s, qn, sn);
    int tot_t = (NQ + NS) * CH * LP;   // 4,902,912
    transpose_cast_kernel<<<(tot_t + 255) / 256, 256, 0, stream>>>(q, s, qT, sT);

    dim3 grid(NQ * NS, JT);   // 800 pairs x 13 row tiles
    // q-direction: A = qn (b), B = sn (k), V = support values (transposed bf16)
    cross_attn_kernel<<<grid, 128, 0, stream>>>(qn, sn, sT, attn_q, recon_q, NS);
    // s-direction: A = sn (k), B = qn (b), V = query values (transposed bf16)
    cross_attn_kernel<<<grid, 128, 0, stream>>>(sn, qn, qT, attn_s, recon_s, NQ);

    cos_loss_kernel<<<(ROWS_Q + ROWS_S + 7) / 8, 256, 0, stream>>>(q, s, recon_q, recon_s, accs);
    finalize_kernel<<<1, 32, 0, stream>>>(accs, losses);
}
